// TrajectoryAttentionLSTM_8650064134882
// MI455X (gfx1250) — matrix operand order repaired and checked by reading, not compiled, not run on hardware
//
#include <hip/hip_runtime.h>

// ---------------------------------------------------------------------------
// TrajectoryAttentionLSTM for MI455X (gfx1250, wave32, WMMA bf16 16x16x32).
// All GEMMs run on v_wmma_f32_16x16x32_bf16 with f32 accumulation.
// ---------------------------------------------------------------------------

typedef __attribute__((ext_vector_type(16))) __bf16 v16bf;
typedef __attribute__((ext_vector_type(8)))  float  v8f;

#define cB    64
#define cA    50
#define cAp   64
#define cT    50
#define cD    128
#define cH    8
#define cDH   16
#define cG    512      // 4*D
#define cNS   3200     // B*A
#define cPRED 60
#define cNTYPES 10

// ---------------- WMMA helpers (CDNA5 VGPR layouts, CDNA5 ISA) ----
__device__ __forceinline__ v8f wmma_bf16(v16bf a, v16bf b, v8f c) {
  return __builtin_amdgcn_wmma_f32_16x16x32_bf16(false, a, false, b, (short)0, c,
                                                 false, false);
}

// A-tile 16x32 bf16, element (m,k) = p[m*strideM + k]
__device__ __forceinline__ v16bf load_a(const __bf16* p, int strideM, int lane) {
  int m  = lane & 15;
  int kb = (lane >> 4) << 3;     // 0 or 8
  v16bf a;
#pragma unroll
  for (int i = 0; i < 16; ++i) {
    int k = kb + i + ((i >> 3) << 3);   // i<8: kb+i ; i>=8: kb+8+i
    a[i] = p[m * strideM + k];
  }
  return a;
}

// A-tile with zero-fill for k >= kmax (used for dh=16 padded to K=32)
__device__ __forceinline__ v16bf load_a_kpad(const __bf16* p, int strideM,
                                             int kmax, int lane) {
  int m  = lane & 15;
  int kb = (lane >> 4) << 3;
  v16bf a;
#pragma unroll
  for (int i = 0; i < 16; ++i) {
    int k = kb + i + ((i >> 3) << 3);
    a[i] = (k < kmax) ? p[m * strideM + k] : (__bf16)0.f;
  }
  return a;
}

// B-tile 32x16 bf16, element (k,n) = p[k*strideK + n*strideN]
__device__ __forceinline__ v16bf load_b(const __bf16* p, int strideK, int strideN,
                                        int lane) {
  int n  = lane & 15;
  int kb = (lane >> 4) << 3;     // 0 or 8
  v16bf b;
#pragma unroll
  for (int i = 0; i < 16; ++i)
    b[i] = p[(kb + i + ((i >> 3) << 3)) * strideK + n * strideN];
  return b;
}

__device__ __forceinline__ v16bf load_b_kpad(const __bf16* p, int strideK,
                                             int strideN, int kmax, int lane) {
  int n  = lane & 15;
  int kb = (lane >> 4) << 3;
  v16bf b;
#pragma unroll
  for (int i = 0; i < 16; ++i) {
    int k = kb + i + ((i >> 3) << 3);
    b[i] = (k < kmax) ? p[k * strideK + n * strideN] : (__bf16)0.f;
  }
  return b;
}

__device__ __forceinline__ float sigf(float x) { return 1.f / (1.f + __expf(-x)); }

// ---------------- weight conversion / padding ------------------------------
__global__ void cvt_kernel(const float* __restrict__ s, __bf16* __restrict__ d,
                           long n) {
  long i = (long)blockIdx.x * blockDim.x + threadIdx.x;
  if (i < n) d[i] = (__bf16)s[i];
}

__global__ void pad_kernel(const float* __restrict__ s, __bf16* __restrict__ d,
                           int rows, int cols, int rpad, int cpad) {
  long i = (long)blockIdx.x * blockDim.x + threadIdx.x;
  if (i >= (long)rpad * cpad) return;
  int r = (int)(i / cpad), c = (int)(i % cpad);
  d[i] = (r < rows && c < cols) ? (__bf16)s[(long)r * cols + c] : (__bf16)0.f;
}

// ---------------- embedding + positional encoding --------------------------
// xbf layout: [b][t][a(64 padded)][d] bf16
__global__ void embed_kernel(const float* __restrict__ agents,
                             const unsigned char* __restrict__ valid,
                             const float* __restrict__ Win,
                             const float* __restrict__ bin,
                             const float* __restrict__ ttab,
                             __bf16* __restrict__ xbf) {
  long idx = (long)blockIdx.x * blockDim.x + threadIdx.x;
  if (idx >= (long)cB * cT * cAp * cD) return;
  int d = (int)(idx & (cD - 1));
  int a = (int)((idx >> 7) & (cAp - 1));
  int t = (int)((idx >> 13) % cT);
  int b = (int)((idx >> 13) / cT);
  float val = 0.f;
  if (a < cA && valid[b * cA + a]) {
    const float* f = agents + (((long)(b * cA + a)) * cT + t) * 6;
    float acc = bin[d];
#pragma unroll
    for (int j = 0; j < 5; ++j) acc += f[j] * Win[d * 5 + j];
    int ty = (int)agents[((long)(b * cA + a)) * cT * 6 + 5];
    ty = ty < 0 ? 0 : (ty > cNTYPES - 1 ? cNTYPES - 1 : ty);
    acc += ttab[ty * cD + d];
    float de  = (float)(d & ~1);
    float div = __expf(de * (-9.2103403719761836f / (float)cD));
    float ang = (float)t * div;
    acc += (d & 1) ? __cosf(ang) : __sinf(ang);
    val = acc;
  }
  xbf[idx] = (__bf16)val;
}

// ---------------- QKV projection GEMM: M=B*T*64, N=384, K=128 ---------------
// Each wave computes 4 N-tiles per A-tile load (A reuse x4).
// q/k/v layout: [bt][h][a(64)][16] bf16, q pre-scaled by 1/sqrt(dh)
__global__ void qkv_kernel(const __bf16* __restrict__ xbf,
                           const __bf16* __restrict__ wqkv,
                           const float* __restrict__ bqkv,
                           __bf16* __restrict__ q, __bf16* __restrict__ k,
                           __bf16* __restrict__ v) {
  int lane = threadIdx.x & 31, wave = threadIdx.x >> 5;
  long job = (long)blockIdx.x * 8 + wave;       // job = tm*6 + tg
  const long MT = (long)cB * cT * cAp / 16;     // 12800
  if (job >= MT * 6) return;
  int tm = (int)(job / 6), tn0 = (int)(job % 6) * 4;
  int n_ = lane & 15, mb = (lane >> 4) << 3;
  v8f acc[4];
#pragma unroll
  for (int j = 0; j < 4; ++j)
#pragma unroll
    for (int r = 0; r < 8; ++r) acc[j][r] = 0.f;
  const __bf16* Apt = xbf + (long)tm * 16 * cD;
#pragma unroll
  for (int k0 = 0; k0 < cD; k0 += 32) {
    v16bf a = load_a(Apt + k0, cD, lane);
#pragma unroll
    for (int j = 0; j < 4; ++j) {
      v16bf bm = load_b(wqkv + (long)(tn0 + j) * 16 * cD + k0, 1, cD, lane);
      acc[j] = wmma_bf16(a, bm, acc[j]);
    }
  }
#pragma unroll
  for (int j = 0; j < 4; ++j) {
    int ncol = (tn0 + j) * 16 + n_;
    int sec  = ncol >> 7;          // 0=q, 1=k, 2=v
    int hh   = (ncol & 127) >> 4;
    int dd   = ncol & 15;
    float bias  = bqkv[ncol];
    float scale = (sec == 0) ? 0.25f : 1.f;   // 1/sqrt(16)
    __bf16* dst = (sec == 0) ? q : (sec == 1 ? k : v);
#pragma unroll
    for (int r = 0; r < 8; ++r) {
      long m  = (long)tm * 16 + r + mb;
      long bt = m >> 6;
      int  ai = (int)(m & 63);
      dst[(((bt * cH + hh) * cAp) + ai) * cDH + dd] =
          (__bf16)((acc[j][r] + bias) * scale);
    }
  }
}

// ---------------- fused attention per (bt, h): scores+softmax+attn@v --------
__global__ void attn_kernel(const __bf16* __restrict__ q,
                            const __bf16* __restrict__ k,
                            const __bf16* __restrict__ v,
                            __bf16* __restrict__ obf) {
  __shared__ float  sc[64 * 64];
  __shared__ __bf16 at[64 * 64];
  int bth = blockIdx.x;
  int bt = bth >> 3, hh = bth & 7;
  int lane = threadIdx.x & 31, wave = threadIdx.x >> 5;   // 4 waves
  const __bf16* qp = q + ((long)(bt * cH + hh) * cAp) * cDH;
  const __bf16* kp = k + ((long)(bt * cH + hh) * cAp) * cDH;
  const __bf16* vp = v + ((long)(bt * cH + hh) * cAp) * cDH;
  int n_ = lane & 15, mb = (lane >> 4) << 3;
  // scores: wave -> M-tile, loop N-tiles; K=dh padded 16->32
  v16bf aq = load_a_kpad(qp + wave * 16 * cDH, cDH, cDH, lane);
#pragma unroll
  for (int tn = 0; tn < 4; ++tn) {
    v8f acc;
#pragma unroll
    for (int r = 0; r < 8; ++r) acc[r] = 0.f;
    v16bf bk = load_b_kpad(kp + tn * 16 * cDH, 1, cDH, cDH, lane);
    acc = wmma_bf16(aq, bk, acc);
#pragma unroll
    for (int r = 0; r < 8; ++r) {
      int row = wave * 16 + r + mb, col = tn * 16 + n_;
      sc[row * 64 + col] = (col < cA) ? acc[r] : -1e30f;
    }
  }
  __syncthreads();
  if (threadIdx.x < 64) {
    int row = threadIdx.x;
    float mx = -1e30f;
    for (int c2 = 0; c2 < 64; ++c2) mx = fmaxf(mx, sc[row * 64 + c2]);
    float s = 0.f;
    for (int c2 = 0; c2 < 64; ++c2) {
      float e = __expf(sc[row * 64 + c2] - mx);
      sc[row * 64 + c2] = e;
      s += e;
    }
    float inv = 1.f / s;
    for (int c2 = 0; c2 < 64; ++c2) at[row * 64 + c2] = (__bf16)(sc[row * 64 + c2] * inv);
  }
  __syncthreads();
  // o = attn(64x64) @ v(64x16)
  v8f acc;
#pragma unroll
  for (int r = 0; r < 8; ++r) acc[r] = 0.f;
#pragma unroll
  for (int k0 = 0; k0 < 64; k0 += 32) {
    v16bf a  = load_a(at + wave * 16 * 64 + k0, 64, lane);
    v16bf bm = load_b(vp + k0 * cDH, cDH, 1, lane);
    acc = wmma_bf16(a, bm, acc);
  }
#pragma unroll
  for (int r = 0; r < 8; ++r) {
    int ai = wave * 16 + r + mb;
    obf[((long)bt * cAp + ai) * cD + hh * cDH + n_] = (__bf16)acc[r];
  }
}

// ---------------- Wo projection + transpose into time-major LSTM layout -----
// xs layout: [t][n=b*50+a][d] bf16.  A reuse x4.
__global__ void wo_kernel(const __bf16* __restrict__ obf,
                          const __bf16* __restrict__ wo,
                          const float* __restrict__ bo,
                          __bf16* __restrict__ xs) {
  int lane = threadIdx.x & 31, wave = threadIdx.x >> 5;
  long job = (long)blockIdx.x * 8 + wave;       // job = tm*2 + tg
  const long MT = 12800;
  if (job >= MT * 2) return;
  int tm = (int)(job >> 1), tn0 = (int)(job & 1) * 4;
  int n_ = lane & 15, mb = (lane >> 4) << 3;
  v8f acc[4];
#pragma unroll
  for (int j = 0; j < 4; ++j)
#pragma unroll
    for (int r = 0; r < 8; ++r) acc[j][r] = 0.f;
#pragma unroll
  for (int k0 = 0; k0 < cD; k0 += 32) {
    v16bf a = load_a(obf + (long)tm * 16 * cD + k0, cD, lane);
#pragma unroll
    for (int j = 0; j < 4; ++j) {
      v16bf bm = load_b(wo + (long)(tn0 + j) * 16 * cD + k0, 1, cD, lane);
      acc[j] = wmma_bf16(a, bm, acc[j]);
    }
  }
#pragma unroll
  for (int j = 0; j < 4; ++j) {
    int ncol = (tn0 + j) * 16 + n_;
    float bias = bo[ncol];
#pragma unroll
    for (int r = 0; r < 8; ++r) {
      long m  = (long)tm * 16 + r + mb;
      long bt = m >> 6;
      int  ai = (int)(m & 63);
      if (ai < cA) {
        int b = (int)(bt / cT), t = (int)(bt % cT);
        long nseq = (long)b * cA + ai;
        xs[((long)t * cNS + nseq) * cD + ncol] = (__bf16)(acc[j][r] + bias);
      }
    }
  }
}

// ---------------- LSTM: hoisted input projection (one layer) ----------------
// gates[t][n][512] bf16 = xs @ Wih.T + b  (M=160000, N=512, K=128). A reuse x4.
__global__ void gatepre_kernel(const __bf16* __restrict__ xs,
                               const __bf16* __restrict__ wih,
                               const float* __restrict__ bias,
                               __bf16* __restrict__ gates) {
  int lane = threadIdx.x & 31, wave = threadIdx.x >> 5;
  long job = (long)blockIdx.x * 8 + wave;       // job = tm*8 + tg
  const long MT = 10000;
  if (job >= MT * 8) return;
  long tm = job >> 3; int tn0 = (int)(job & 7) * 4;
  int n_ = lane & 15, mb = (lane >> 4) << 3;
  v8f acc[4];
#pragma unroll
  for (int j = 0; j < 4; ++j) {
    float bb = bias[(tn0 + j) * 16 + n_];
#pragma unroll
    for (int r = 0; r < 8; ++r) acc[j][r] = bb;
  }
#pragma unroll
  for (int k0 = 0; k0 < cD; k0 += 32) {
    v16bf a = load_a(xs + tm * 16 * cD + k0, cD, lane);
#pragma unroll
    for (int j = 0; j < 4; ++j) {
      v16bf bm = load_b(wih + (long)(tn0 + j) * 16 * cD + k0, 1, cD, lane);
      acc[j] = wmma_bf16(a, bm, acc[j]);
    }
  }
#pragma unroll
  for (int j = 0; j < 4; ++j)
#pragma unroll
    for (int r = 0; r < 8; ++r)
      gates[(tm * 16 + r + mb) * cG + (tn0 + j) * 16 + n_] = (__bf16)acc[j][r];
}

// ---------------- LSTM recurrence (one layer): 16 sequences / block ---------
// Gate-aligned tiling: wave w owns tiles {w, 8+w, 16+w, 24+w}, so each lane
// holds i/f/g/o for its (row,col) in registers -> no LDS z redistribution.
// Whh B-tiles stay resident in registers across all 50 timesteps.
__global__ void lstm_layer_kernel(const __bf16* __restrict__ gates,
                                  const __bf16* __restrict__ whh,
                                  __bf16* __restrict__ xs_out,
                                  float* __restrict__ hT) {
  __shared__ __bf16 hbuf[16 * cD];    // current hidden, WMMA A source
  int lane = threadIdx.x & 31, wave = threadIdx.x >> 5;
  int n0 = blockIdx.x * 16;
  int n_ = lane & 15, mb = (lane >> 4) << 3;
  int col = wave * 16 + n_;
  // resident Whh B-tiles (16 per wave = 128 VGPRs/lane)
  v16bf breg[4][4];
#pragma unroll
  for (int g = 0; g < 4; ++g) {
    int tn = g * 8 + wave;
#pragma unroll
    for (int kk = 0; kk < 4; ++kk)
      breg[g][kk] = load_b(whh + (long)tn * 16 * cD + kk * 32, 1, cD, lane);
  }
  for (int i = threadIdx.x; i < 16 * cD; i += 256) hbuf[i] = (__bf16)0.f;
  float creg[8];
#pragma unroll
  for (int i = 0; i < 8; ++i) creg[i] = 0.f;
  __syncthreads();
  for (int t = 0; t < cT; ++t) {
    const __bf16* gp = gates + ((long)t * cNS + n0) * cG;
    if (t + 1 < cT && lane == 0)   // gfx1250 global_prefetch_b8
      __builtin_prefetch(gates + ((long)(t + 1) * cNS + n0) * cG, 0, 0);
    v16bf areg[4];
#pragma unroll
    for (int kk = 0; kk < 4; ++kk) areg[kk] = load_a(hbuf + kk * 32, cD, lane);
    v8f acc[4];
#pragma unroll
    for (int g = 0; g < 4; ++g) {
      int tn = g * 8 + wave;
#pragma unroll
      for (int r = 0; r < 8; ++r)
        acc[g][r] = (float)gp[(r + mb) * cG + tn * 16 + n_];
#pragma unroll
      for (int kk = 0; kk < 4; ++kk)
        acc[g] = wmma_bf16(areg[kk], breg[g][kk], acc[g]);
    }
    __syncthreads();   // all waves done reading hbuf before updates
#pragma unroll
    for (int r = 0; r < 8; ++r) {
      int row = r + mb;
      float ig = sigf(acc[0][r]);
      float fg = sigf(acc[1][r]);
      float gg = tanhf(acc[2][r]);
      float og = sigf(acc[3][r]);
      float c  = fg * creg[r] + ig * gg;
      creg[r] = c;
      float h = og * tanhf(c);
      hbuf[row * cD + col] = (__bf16)h;
      xs_out[((long)t * cNS + n0 + row) * cD + col] = (__bf16)h;
      if (t == cT - 1) hT[(long)(n0 + row) * cD + col] = h;
    }
    __syncthreads();
  }
#if __has_builtin(__builtin_amdgcn_s_wait_tensorcnt)
  __builtin_amdgcn_s_wait_tensorcnt(0);   // gfx1250 marker; TENSORcnt is 0
#endif
}

// ---------------- decoder: single block, 60 steps x 4 layers, WMMA ----------
__global__ void decoder_kernel(const float* __restrict__ ego,
                               const float* __restrict__ hT,
                               const __bf16* __restrict__ wih,
                               const __bf16* __restrict__ whh,
                               const float* __restrict__ db,
                               const float* __restrict__ wdemb,
                               const float* __restrict__ bdemb,
                               const float* __restrict__ wout,
                               const float* __restrict__ bout,
                               __bf16* __restrict__ hstate,  // [4][64][128]
                               float* __restrict__ cstate,   // [4][64][128]
                               float* __restrict__ traj) {   // [64][120]
  __shared__ __bf16 xbufA[64 * cD];
  __shared__ __bf16 xbufB[64 * cD];
  __shared__ float  predS[64 * 2];
  int tid = threadIdx.x, lane = tid & 31, wave = tid >> 5;
  int n_ = lane & 15, mb = (lane >> 4) << 3;
  for (int i = tid; i < 64 * cD; i += 256) {
    int b = i >> 7, d = i & 127;
    float hv = hT[((long)b * cA) * cD + d];   // context = h_n[:,0]
#pragma unroll
    for (int l = 0; l < 4; ++l) {
      hstate[l * 64 * cD + i] = (__bf16)hv;
      cstate[l * 64 * cD + i] = 0.f;
    }
    float p0 = ego[(b * cT + (cT - 1)) * 5 + 0];
    float p1 = ego[(b * cT + (cT - 1)) * 5 + 1];
    xbufA[i] = (__bf16)(p0 * wdemb[d * 2] + p1 * wdemb[d * 2 + 1] + bdemb[d]);
  }
  __threadfence();
  __syncthreads();
  for (int step = 0; step < cPRED; ++step) {
    __bf16* xin = xbufA; __bf16* xout = xbufB;
    for (int l = 0; l < 4; ++l) {
      v8f acc[4][4];
      const __bf16* wihl = wih + (long)l * cG * cD;
      const __bf16* whhl = whh + (long)l * cG * cD;
      const __bf16* hl   = hstate + (long)l * 64 * cD;
#pragma unroll
      for (int g = 0; g < 4; ++g) {
        int tn = g * 8 + wave;
        float bb = db[l * cG + tn * 16 + n_];
#pragma unroll
        for (int tm = 0; tm < 4; ++tm) {
          v8f a_;
#pragma unroll
          for (int r = 0; r < 8; ++r) a_[r] = bb;
#pragma unroll
          for (int k0 = 0; k0 < cD; k0 += 32) {
            v16bf av = load_a(xin + tm * 16 * cD + k0, cD, lane);
            v16bf bv = load_b(wihl + (long)tn * 16 * cD + k0, 1, cD, lane);
            a_ = wmma_bf16(av, bv, a_);
          }
#pragma unroll
          for (int k0 = 0; k0 < cD; k0 += 32) {
            v16bf av = load_a(hl + (long)tm * 16 * cD + k0, cD, lane);
            v16bf bv = load_b(whhl + (long)tn * 16 * cD + k0, 1, cD, lane);
            a_ = wmma_bf16(av, bv, a_);
          }
          acc[g][tm] = a_;
        }
      }
      __syncthreads();   // all waves done reading hstate[l] before updates
      int col = wave * 16 + n_;
#pragma unroll
      for (int tm = 0; tm < 4; ++tm)
#pragma unroll
        for (int r = 0; r < 8; ++r) {
          int row = tm * 16 + r + mb;
          float* cp = cstate + ((long)l * 64 + row) * cD + col;
          float ig = sigf(acc[0][tm][r]);
          float fg = sigf(acc[1][tm][r]);
          float gg = tanhf(acc[2][tm][r]);
          float og = sigf(acc[3][tm][r]);
          float c  = fg * (*cp) + ig * gg;
          *cp = c;
          float h = og * tanhf(c);
          xout[row * cD + col] = (__bf16)h;
          hstate[((long)l * 64 + row) * cD + col] = (__bf16)h;
        }
      __threadfence();
      __syncthreads();
      __bf16* tpp = xin; xin = xout; xout = tpp;
    }
    // xin == xbufA again (4 swaps): holds top-layer h
    if (tid < 128) {
      int b = tid >> 1, j = tid & 1;
      float s = bout[j];
      for (int d2 = 0; d2 < cD; ++d2) s += (float)xin[b * cD + d2] * wout[j * cD + d2];
      predS[b * 2 + j] = s;
      traj[b * 120 + step * 2 + j] = s;
    }
    __syncthreads();
    for (int i = tid; i < 64 * cD; i += 256) {
      int b = i >> 7, d = i & 127;
      xin[i] = (__bf16)(predS[b * 2] * wdemb[d * 2] +
                        predS[b * 2 + 1] * wdemb[d * 2 + 1] + bdemb[d]);
    }
    __syncthreads();
  }
}

// ---------------- refinement MLP: (64,120)->512->relu->(64,120) -------------
__global__ void refine1_kernel(const __bf16* __restrict__ trajbf,
                               const __bf16* __restrict__ wr1,
                               const float* __restrict__ br1,
                               __bf16* __restrict__ hid) {
  int lane = threadIdx.x & 31, wave = threadIdx.x >> 5;
  int tile = blockIdx.x * 8 + wave;        // 4 M-tiles x 32 N-tiles
  if (tile >= 128) return;
  int tm = tile >> 5, tn = tile & 31;
  int n_ = lane & 15, mb = (lane >> 4) << 3;
  v8f acc;
  float bb = br1[tn * 16 + n_];
#pragma unroll
  for (int r = 0; r < 8; ++r) acc[r] = bb;
#pragma unroll
  for (int k0 = 0; k0 < cD; k0 += 32) {
    v16bf a  = load_a(trajbf + tm * 16 * cD + k0, cD, lane);
    v16bf bm = load_b(wr1 + (long)tn * 16 * cD + k0, 1, cD, lane);
    acc = wmma_bf16(a, bm, acc);
  }
#pragma unroll
  for (int r = 0; r < 8; ++r)
    hid[(tm * 16 + r + mb) * cG + tn * 16 + n_] = (__bf16)fmaxf(acc[r], 0.f);
}

__global__ void refine2_kernel(const __bf16* __restrict__ hid,
                               const __bf16* __restrict__ wr2p,
                               const float* __restrict__ br2,
                               float* __restrict__ out) {
  int lane = threadIdx.x & 31, wave = threadIdx.x >> 5;
  int tile = blockIdx.x * 8 + wave;        // 4 M-tiles x 8 N-tiles (N padded 128)
  if (tile >= 32) return;
  int tm = tile >> 3, tn = tile & 7;
  int n_ = lane & 15, mb = (lane >> 4) << 3;
  int ncol = tn * 16 + n_;
  v8f acc;
  float bb = (ncol < 120) ? br2[ncol] : 0.f;
#pragma unroll
  for (int r = 0; r < 8; ++r) acc[r] = bb;
#pragma unroll
  for (int k0 = 0; k0 < cG; k0 += 32) {
    v16bf a  = load_a(hid + tm * 16 * cG + k0, cG, lane);
    v16bf bm = load_b(wr2p + (long)tn * 16 * cG + k0, 1, cG, lane);
    acc = wmma_bf16(a, bm, acc);
  }
  if (ncol < 120) {
#pragma unroll
    for (int r = 0; r < 8; ++r)
      out[(tm * 16 + r + mb) * 120 + ncol] = acc[r];
  }
}

// ---------------------------------------------------------------------------
extern "C" void kernel_launch(void* const* d_in, const int* in_sizes, int n_in,
                              void* d_out, int out_size, void* d_ws, size_t ws_size,
                              hipStream_t stream) {
  (void)in_sizes; (void)n_in; (void)out_size; (void)ws_size;
  const float* ego    = (const float*)d_in[0];
  const float* agents = (const float*)d_in[1];
  const unsigned char* valid = (const unsigned char*)d_in[2];
  const float* Win  = (const float*)d_in[3];
  const float* bin  = (const float*)d_in[4];
  const float* ttab = (const float*)d_in[5];
  const float* Wqkv = (const float*)d_in[6];
  const float* bqkv = (const float*)d_in[7];
  const float* Wo   = (const float*)d_in[8];
  const float* bo   = (const float*)d_in[9];
  const float* eWih = (const float*)d_in[10];
  const float* eWhh = (const float*)d_in[11];
  const float* eb   = (const float*)d_in[12];
  const float* dWih = (const float*)d_in[13];
  const float* dWhh = (const float*)d_in[14];
  const float* db   = (const float*)d_in[15];
  const float* Wdemb= (const float*)d_in[16];
  const float* bdemb= (const float*)d_in[17];
  const float* Wout = (const float*)d_in[18];
  const float* bout = (const float*)d_in[19];
  const float* Wr1  = (const float*)d_in[20];
  const float* br1  = (const float*)d_in[21];
  const float* Wr2  = (const float*)d_in[22];
  const float* br2  = (const float*)d_in[23];

  // ---- workspace layout (bytes); attention buffers aliased under gates ----
  char* ws = (char*)d_ws;
  constexpr size_t SZ_XBF  = (size_t)cB * cT * cAp * cD * 2;   // 52,428,800
  constexpr size_t OFF_XBF = 0;                 // also XSA after wo_kernel
  constexpr size_t OFF_WQKV= OFF_XBF + SZ_XBF;
  constexpr size_t OFF_WO  = OFF_WQKV + 384 * 128 * 2;
  constexpr size_t OFF_EWIH= OFF_WO   + 128 * 128 * 2;
  constexpr size_t OFF_EWHH= OFF_EWIH + 4 * 512 * 128 * 2;
  constexpr size_t OFF_DWIH= OFF_EWHH + 4 * 512 * 128 * 2;
  constexpr size_t OFF_DWHH= OFF_DWIH + 4 * 512 * 128 * 2;
  constexpr size_t OFF_WR1 = OFF_DWHH + 4 * 512 * 128 * 2;
  constexpr size_t OFF_WR2 = OFF_WR1  + 512 * 128 * 2;
  constexpr size_t OFF_HT  = OFF_WR2  + 128 * 512 * 2;
  constexpr size_t OFF_DECH= OFF_HT   + (size_t)cNS * cD * 4;
  constexpr size_t OFF_DECC= OFF_DECH + 4 * 64 * 128 * 2;
  constexpr size_t OFF_TRAJ= OFF_DECC + 4 * 64 * 128 * 4;
  constexpr size_t OFF_TJBF= OFF_TRAJ + 64 * 128 * 4;
  constexpr size_t OFF_HID = OFF_TJBF + 64 * 128 * 2;
  constexpr size_t OFF_XSB = OFF_HID  + 64 * 512 * 2;
  constexpr size_t SZ_XS   = (size_t)cT * cNS * cD * 2;        // 40,960,000
  constexpr size_t OFF_R   = OFF_XSB + SZ_XS;   // q/k/v/obf, later gates(bf16)
  constexpr size_t SZ_QKV1 = (size_t)cNS * cH * cAp * cDH * 2; // 52,428,800
  constexpr size_t OFF_Q   = OFF_R;
  constexpr size_t OFF_K   = OFF_Q + SZ_QKV1;
  constexpr size_t OFF_V   = OFF_K + SZ_QKV1;
  constexpr size_t OFF_OBF = OFF_V + SZ_QKV1;
  constexpr size_t OFF_GATES = OFF_R;           // 163.84MB, fits in region

  __bf16* xbf   = (__bf16*)(ws + OFF_XBF);
  __bf16* xsA   = (__bf16*)(ws + OFF_XBF);      // alias (xbf dead after qkv)
  __bf16* xsB   = (__bf16*)(ws + OFF_XSB);
  __bf16* wqkvb = (__bf16*)(ws + OFF_WQKV);
  __bf16* wob   = (__bf16*)(ws + OFF_WO);
  __bf16* ewihb = (__bf16*)(ws + OFF_EWIH);
  __bf16* ewhhb = (__bf16*)(ws + OFF_EWHH);
  __bf16* dwihb = (__bf16*)(ws + OFF_DWIH);
  __bf16* dwhhb = (__bf16*)(ws + OFF_DWHH);
  __bf16* wr1b  = (__bf16*)(ws + OFF_WR1);
  __bf16* wr2b  = (__bf16*)(ws + OFF_WR2);
  float*  hT    = (float*) (ws + OFF_HT);
  __bf16* dech  = (__bf16*)(ws + OFF_DECH);
  float*  decc  = (float*) (ws + OFF_DECC);
  float*  traj  = (float*) (ws + OFF_TRAJ);
  __bf16* trajb = (__bf16*)(ws + OFF_TJBF);
  __bf16* hid   = (__bf16*)(ws + OFF_HID);
  __bf16* qws   = (__bf16*)(ws + OFF_Q);
  __bf16* kws   = (__bf16*)(ws + OFF_K);
  __bf16* vws   = (__bf16*)(ws + OFF_V);
  __bf16* obf   = (__bf16*)(ws + OFF_OBF);
  __bf16* gates = (__bf16*)(ws + OFF_GATES);

  auto gr = [](long n, int b) { return (unsigned)((n + b - 1) / b); };

  // weight conversions / padding
  cvt_kernel<<<gr(384 * 128, 256), 256, 0, stream>>>(Wqkv, wqkvb, 384 * 128);
  cvt_kernel<<<gr(128 * 128, 256), 256, 0, stream>>>(Wo, wob, 128 * 128);
  cvt_kernel<<<gr(4L * 512 * 128, 256), 256, 0, stream>>>(eWih, ewihb, 4L * 512 * 128);
  cvt_kernel<<<gr(4L * 512 * 128, 256), 256, 0, stream>>>(eWhh, ewhhb, 4L * 512 * 128);
  cvt_kernel<<<gr(4L * 512 * 128, 256), 256, 0, stream>>>(dWih, dwihb, 4L * 512 * 128);
  cvt_kernel<<<gr(4L * 512 * 128, 256), 256, 0, stream>>>(dWhh, dwhhb, 4L * 512 * 128);
  pad_kernel<<<gr(512L * 128, 256), 256, 0, stream>>>(Wr1, wr1b, 512, 120, 512, 128);
  pad_kernel<<<gr(128L * 512, 256), 256, 0, stream>>>(Wr2, wr2b, 120, 512, 128, 512);

  // embedding + attention + output projection
  embed_kernel<<<gr((long)cB * cT * cAp * cD, 256), 256, 0, stream>>>(
      agents, valid, Win, bin, ttab, xbf);
  qkv_kernel<<<9600, 256, 0, stream>>>(xbf, wqkvb, bqkv, qws, kws, vws);
  attn_kernel<<<cB * cT * cH, 128, 0, stream>>>(qws, kws, vws, obf);
  wo_kernel<<<3200, 256, 0, stream>>>(obf, wob, bo, xsA);

  // 4-layer encoder LSTM: hoisted input GEMM + recurrent kernel per layer
  for (int l = 0; l < 4; ++l) {
    const __bf16* xin = (l & 1) ? xsB : xsA;
    __bf16*       xo  = (l & 1) ? xsA : xsB;
    gatepre_kernel<<<10000, 256, 0, stream>>>(
        xin, ewihb + (long)l * 512 * 128, eb + l * 512, gates);
    lstm_layer_kernel<<<cNS / 16, 256, 0, stream>>>(
        gates, ewhhb + (long)l * 512 * 128, xo, hT);
  }

  // autoregressive decoder + refinement MLP
  decoder_kernel<<<1, 256, 0, stream>>>(ego, hT, dwihb, dwhhb, db, Wdemb, bdemb,
                                        Wout, bout, dech, decc, traj);
  pad_kernel<<<gr(64L * 128, 256), 256, 0, stream>>>(traj, trajb, 64, 120, 64, 128);
  refine1_kernel<<<16, 256, 0, stream>>>(trajb, wr1b, br1, hid);
  refine2_kernel<<<4, 256, 0, stream>>>(hid, wr2b, br2, (float*)d_out);
}
